// ElectronicEmbedding_13005160972680
// MI455X (gfx1250) — matrix operand order repaired and checked by reading, not compiled, not run on hardware
//
#include <hip/hip_runtime.h>

typedef __attribute__((ext_vector_type(16))) __bf16 v16bf;
typedef __attribute__((ext_vector_type(8)))  __bf16 v8bf;
typedef __attribute__((ext_vector_type(8)))  float  v8f;
typedef __attribute__((ext_vector_type(4)))  float  v4f;
typedef __attribute__((ext_vector_type(4)))  unsigned int v4u;
typedef __attribute__((ext_vector_type(8)))  int    v8i;
typedef __attribute__((ext_vector_type(4)))  int    v4i;

union BF16x16 { v16bf v; v8bf h[2]; };

#if defined(__gfx1250__) && __has_builtin(__builtin_amdgcn_tensor_load_to_lds) && \
    __has_builtin(__builtin_amdgcn_s_wait_tensorcnt)
#define USE_TDM 1
#endif

__device__ __forceinline__ float swishf(float x, float alpha, float beta) {
    // alpha * x * sigmoid(beta * x)
    return alpha * x / (1.0f + __expf(-beta * x));
}

// ---------------------------------------------------------------------------
// 1) Transpose + convert a 256x256 fp32 weight to bf16, Wt[j*256+k] = W[k*256+j]
// ---------------------------------------------------------------------------
__global__ __launch_bounds__(256) void convert_w(const float* __restrict__ W,
                                                 unsigned short* __restrict__ Wt) {
    for (int idx = blockIdx.x * 256 + threadIdx.x; idx < 256 * 256; idx += gridDim.x * 256) {
        int j = idx >> 8, k = idx & 255;
        __bf16 b = (__bf16)W[k * 256 + j];
        Wt[idx] = __builtin_bit_cast(unsigned short, b);
    }
}

// ---------------------------------------------------------------------------
// 2) kp2 = W_lin @ k_plus, km2 = W_lin @ k_minus, bk = {b.k_plus, b.k_minus}
//    (the entire first GEMM of the reference folds into these two mat-vecs)
// ---------------------------------------------------------------------------
__global__ __launch_bounds__(256) void kvec_kernel(const float* __restrict__ W_lin,
                                                   const float* __restrict__ b_lin,
                                                   const float* __restrict__ kp,
                                                   const float* __restrict__ km,
                                                   float* __restrict__ kp2,
                                                   float* __restrict__ km2,
                                                   float* __restrict__ bk) {
    int d = threadIdx.x;
    float sp = 0.f, sm = 0.f;
    for (int j = 0; j < 256; ++j) {
        float w = W_lin[d * 256 + j];
        sp += w * kp[j];
        sm += w * km[j];
    }
    kp2[d] = sp;
    km2[d] = sm;
    if (d == 0) {
        float a = 0.f, b = 0.f;
        for (int j = 0; j < 256; ++j) { a += b_lin[j] * kp[j]; b += b_lin[j] * km[j]; }
        bk[0] = a; bk[1] = b;
    }
}

// ---------------------------------------------------------------------------
// 3) Exclusive scan of num_atoms + per-atom segment ids (ragged-safe)
// ---------------------------------------------------------------------------
__global__ __launch_bounds__(256) void seg_kernel(const int* __restrict__ num_atoms,
                                                  int Mm, int Nn,
                                                  int* __restrict__ offs,
                                                  int* __restrict__ seg) {
    if (threadIdx.x == 0) {
        int acc = 0;
        for (int m = 0; m < Mm; ++m) { offs[m] = acc; acc += num_atoms[m]; }
        offs[Mm] = acc;
    }
    __syncthreads();
    for (int m = threadIdx.x; m < Mm; m += blockDim.x) {
        int st = offs[m], cnt = num_atoms[m];
        for (int j = 0; j < cnt; ++j) {
            int i = st + j;
            if (i < Nn) seg[i] = m;
        }
    }
}

// ---------------------------------------------------------------------------
// 4) num[i] = softplus((e_z[i] . k2 + b.k) / sqrt(D)); wave32 per atom.
//    e_z is streamed exactly once -> non-temporal loads keep it out of L2.
// ---------------------------------------------------------------------------
__global__ __launch_bounds__(256) void num_kernel(const float* __restrict__ e_z,
                                                  const int* __restrict__ seg,
                                                  const float* __restrict__ psi,
                                                  const float* __restrict__ kp2,
                                                  const float* __restrict__ km2,
                                                  const float* __restrict__ bk,
                                                  float* __restrict__ numv, int Nn) {
    __shared__ float kpS[256], kmS[256];
    int tid = threadIdx.x;
    kpS[tid] = kp2[tid];
    kmS[tid] = km2[tid];
    __syncthreads();
    int lane = tid & 31, wid = tid >> 5;
    int i = blockIdx.x * 8 + wid;
    if (i >= Nn) return;
    if (i + 8 < Nn) __builtin_prefetch(e_z + (size_t)(i + 8) * 256 + lane * 8, 0, 0);
    int s = seg[i];
    bool pos = psi[s] >= 0.f;
    const float* k2 = pos ? kpS : kmS;
    float bias = pos ? bk[0] : bk[1];
    const v4f* row = (const v4f*)(e_z + (size_t)i * 256);
    v4f x0 = __builtin_nontemporal_load(row + lane);
    v4f x1 = __builtin_nontemporal_load(row + lane + 32);
    int c0 = lane * 4, c1 = 128 + lane * 4;
    float sum = x0[0] * k2[c0] + x0[1] * k2[c0 + 1] + x0[2] * k2[c0 + 2] + x0[3] * k2[c0 + 3]
              + x1[0] * k2[c1] + x1[1] * k2[c1 + 1] + x1[2] * k2[c1 + 2] + x1[3] * k2[c1 + 3];
    for (int m = 16; m >= 1; m >>= 1) sum += __shfl_xor(sum, m, 32);
    if (lane == 0) {
        float arg = (sum + bias) * 0.0625f;          // 1/sqrt(256)
        float sp = (arg > 20.f) ? arg : log1pf(__expf(arg));
        numv[i] = sp;
    }
}

// ---------------------------------------------------------------------------
// 5) denom[m] = segment_sum(num); wave32 per molecule
// ---------------------------------------------------------------------------
__global__ __launch_bounds__(256) void denom_kernel(const float* __restrict__ numv,
                                                    const int* __restrict__ offs,
                                                    const int* __restrict__ num_atoms,
                                                    float* __restrict__ denom, int Mm) {
    int lane = threadIdx.x & 31, wid = threadIdx.x >> 5;
    int m = blockIdx.x * 8 + wid;
    if (m >= Mm) return;
    int st = offs[m], cnt = num_atoms[m];
    float s = 0.f;
    for (int j = lane; j < cnt; j += 32) s += numv[st + j];
    for (int d = 16; d >= 1; d >>= 1) s += __shfl_xor(s, d, 32);
    if (lane == 0) denom[m] = s;
}

// ---------------------------------------------------------------------------
// 6) Fused 3-GEMM ResidualMLP chain with bf16 WMMA (f32 accumulate).
//    128-row tile, 512 threads = 16 waves; wave (w&7) -> M stripe, (w>>3) -> N half.
//    Weights staged to LDS via the Tensor Data Mover (TDM) when available.
// ---------------------------------------------------------------------------
#define TM 128
#define WS_LDS_OFF 65536u   // byte offset of Ws inside dynamic LDS block

__device__ __forceinline__ void stage_weights_tdm(const unsigned short* Wg) {
#ifdef USE_TDM
    unsigned long long ga = (unsigned long long)(uintptr_t)Wg;
    v4u g0;
    g0[0] = 1u;                                            // count=1, user descriptor
    g0[1] = WS_LDS_OFF;                                    // lds_addr (bytes)
    g0[2] = (unsigned int)(ga & 0xFFFFFFFFu);              // global_addr[31:0]
    g0[3] = (unsigned int)((ga >> 32) & 0x01FFFFFFu)       // global_addr[56:32]
            | (2u << 30);                                  // type=2 ("image")
    v8i g1;
    g1[0] = 0x00010000;            // wg_mask=0, data_size=1 (2 bytes), no pad/iterate
    g1[1] = (int)(256u << 16);     // atomic_barrier_addr=0 ; tensor_dim0[15:0]=256
    g1[2] = (int)(256u << 16);     // tensor_dim0[31:16]=0 ; tensor_dim1[15:0]=256
    g1[3] = (int)(256u << 16);     // tensor_dim1[31:16]=0 ; tile_dim0=256
    g1[4] = 256;                   // tile_dim1=256 ; tile_dim2=0
    g1[5] = 256;                   // tensor_dim0_stride[31:0]=256
    g1[6] = 0;                     // tensor_dim0_stride[47:32]=0 ; dim1_stride[15:0]=0
    g1[7] = 1;                     // tensor_dim1_stride[47:16]=1 -> stride=65536
    v4i z4 = {0, 0, 0, 0};
#if __clang_major__ >= 23
    v8i z8 = {0, 0, 0, 0, 0, 0, 0, 0};
    __builtin_amdgcn_tensor_load_to_lds(g0, g1, z4, z4, z8, 0);
#else
    __builtin_amdgcn_tensor_load_to_lds(g0, g1, z4, z4, 0);
#endif
    __builtin_amdgcn_s_wait_tensorcnt(0);
#else
    (void)Wg;
#endif
}

__global__ __launch_bounds__(512) void mlp_kernel(const float* __restrict__ numv,
                                                  const float* __restrict__ denom,
                                                  const int* __restrict__ seg,
                                                  const float* __restrict__ psi,
                                                  const float* __restrict__ vplus,
                                                  const float* __restrict__ vminus,
                                                  const float* __restrict__ act_a,
                                                  const float* __restrict__ act_b,
                                                  const unsigned short* __restrict__ W1t,
                                                  const unsigned short* __restrict__ W2t,
                                                  const unsigned short* __restrict__ W3t,
                                                  float* __restrict__ out, int Nn) {
    extern __shared__ char smem[];
    __bf16* Xs = (__bf16*)smem;                                   // 128*256 bf16 = 64KB
    __bf16* Ws = (__bf16*)(smem + WS_LDS_OFF);                    // 256*256 bf16 = 128KB
    float* vpS = (float*)(smem + WS_LDS_OFF + 131072);            // 256 f
    float* vmS = vpS + 256;                                       // 256 f
    float* aRow = vmS + 256;                                      // 128 f
    int* posRow = (int*)(aRow + TM);                              // 128 i

    const int tid = threadIdx.x;
    const int rowBase = blockIdx.x * TM;

    for (int c = tid; c < 256; c += 512) { vpS[c] = vplus[c]; vmS[c] = vminus[c]; }
    if (tid < TM) {
        int i = rowBase + tid;
        if (i >= Nn) i = Nn - 1;
        int s = seg[i];
        float p = psi[s];
        aRow[tid] = p * numv[i] / denom[s];
        posRow[tid] = (p >= 0.f) ? 1 : 0;
    }
    float al0 = act_a[0], be0 = act_b[0];
    float al1 = act_a[1], be1 = act_b[1];
    float al2 = act_a[2], be2 = act_b[2];
    __syncthreads();

    // Stage 0 input: X0 = bf16(swish1(a_i * v_sign))
    for (int idx = tid; idx < TM * 256; idx += 512) {
        int r = idx >> 8, c = idx & 255;
        float v = posRow[r] ? vpS[c] : vmS[c];
        Xs[idx] = (__bf16)swishf(aRow[r] * v, al0, be0);
    }

    const int lane = tid & 31;
    const int wave = tid >> 5;
    const int mBase = (wave & 7) * 16;
    const int nBase = (wave >> 3) * 128;
    const int half = lane >> 4;
    const int l16 = lane & 15;

    v8f acc[8];

    for (int stage = 0; stage < 3; ++stage) {
        const unsigned short* Wg = (stage == 0) ? W1t : ((stage == 1) ? W2t : W3t);

        __syncthreads();  // previous stage fully done with Ws/Xs
#ifdef USE_TDM
        if (tid < 32) {                 // one wave issues the DMA; TDM ignores EXEC
            stage_weights_tdm(Wg);      // tensor_load_to_lds + s_wait_tensorcnt 0
        }
#else
        for (int idx = tid * 8; idx < 256 * 256; idx += 512 * 8) {
            *(v8bf*)(Ws + idx) = *(const v8bf*)((const __bf16*)Wg + idx);
        }
#endif
#pragma unroll
        for (int t = 0; t < 8; ++t)
#pragma unroll
            for (int e = 0; e < 8; ++e) acc[t][e] = 0.f;
        __syncthreads();  // Ws visible to all waves

        // GEMM: C[128x256] += X[128x256] * Wt-tiles, K in chunks of 32
        for (int kc = 0; kc < 8; ++kc) {
            BF16x16 afrag;
            const __bf16* ap = Xs + (size_t)(mBase + l16) * 256 + kc * 32 + half * 16;
            afrag.h[0] = *(const v8bf*)ap;           // K = kc*32 + half*16 .. +8
            afrag.h[1] = *(const v8bf*)(ap + 8);     // K = kc*32 + half*16 + 8 .. +8
            v16bf bfrag[8];
#pragma unroll
            for (int nt = 0; nt < 8; ++nt) {
                int j = nBase + nt * 16 + l16;
                bfrag[nt] = *(const v16bf*)(Ws + (size_t)j * 256 + kc * 32 + half * 16);
            }
#pragma unroll
            for (int nt = 0; nt < 8; ++nt) {
                acc[nt] = __builtin_amdgcn_wmma_f32_16x16x32_bf16(
                    false, afrag.v, false, bfrag[nt], (short)0, acc[nt], false, false);
            }
        }

        __syncthreads();  // all reads of Xs complete before overwrite

        if (stage < 2) {
#pragma unroll
            for (int nt = 0; nt < 8; ++nt) {
#pragma unroll
                for (int e = 0; e < 8; ++e) {
                    int r = mBase + e + half * 8;
                    int c = nBase + nt * 16 + l16;
                    float h = acc[nt][e];
                    float x;
                    if (stage == 0) {
                        x = swishf(h, al1, be1);                       // swish2(h1)
                    } else {
                        float av = aRow[r] * (posRow[r] ? vpS[c] : vmS[c]);
                        x = swishf(av + h, al2, be2);                  // swish3(av + h2)
                    }
                    Xs[r * 256 + c] = (__bf16)x;
                }
            }
        } else {
#pragma unroll
            for (int nt = 0; nt < 8; ++nt) {
#pragma unroll
                for (int e = 0; e < 8; ++e) {
                    int r = mBase + e + half * 8;
                    int c = nBase + nt * 16 + l16;
                    size_t gr = (size_t)(rowBase + r);
                    if (gr < (size_t)Nn)
                        __builtin_nontemporal_store(acc[nt][e], &out[gr * 256 + c]);
                }
            }
        }
    }
}

// ---------------------------------------------------------------------------
extern "C" void kernel_launch(void* const* d_in, const int* in_sizes, int n_in,
                              void* d_out, int out_size, void* d_ws, size_t ws_size,
                              hipStream_t stream) {
    const float* psi       = (const float*)d_in[0];
    const float* e_z       = (const float*)d_in[1];
    const int*   num_atoms = (const int*)d_in[2];
    const float* W_lin     = (const float*)d_in[3];
    const float* b_lin     = (const float*)d_in[4];
    const float* k_plus    = (const float*)d_in[5];
    const float* k_minus   = (const float*)d_in[6];
    const float* v_plus    = (const float*)d_in[7];
    const float* v_minus   = (const float*)d_in[8];
    const float* W_res1    = (const float*)d_in[9];
    const float* W_res2    = (const float*)d_in[10];
    const float* W_out     = (const float*)d_in[11];
    const float* act_alpha = (const float*)d_in[12];
    const float* act_beta  = (const float*)d_in[13];
    float* out = (float*)d_out;

    const int Mm = in_sizes[0];
    const int Nn = in_sizes[1] / 256;

    char* ws = (char*)d_ws;
    size_t off = 0;
    auto alloc = [&](size_t bytes) -> char* {
        char* p = ws + off;
        off += (bytes + 255) & ~(size_t)255;
        return p;
    };
    float* kp2            = (float*)alloc(256 * sizeof(float));
    float* km2            = (float*)alloc(256 * sizeof(float));
    float* bk             = (float*)alloc(2 * sizeof(float));
    int*   offs           = (int*)alloc((size_t)(Mm + 1) * sizeof(int));
    int*   seg            = (int*)alloc((size_t)Nn * sizeof(int));
    float* numv           = (float*)alloc((size_t)Nn * sizeof(float));
    float* denom          = (float*)alloc((size_t)Mm * sizeof(float));
    unsigned short* W1t   = (unsigned short*)alloc(256 * 256 * sizeof(unsigned short));
    unsigned short* W2t   = (unsigned short*)alloc(256 * 256 * sizeof(unsigned short));
    unsigned short* W3t   = (unsigned short*)alloc(256 * 256 * sizeof(unsigned short));

    convert_w<<<64, 256, 0, stream>>>(W_res1, W1t);
    convert_w<<<64, 256, 0, stream>>>(W_res2, W2t);
    convert_w<<<64, 256, 0, stream>>>(W_out, W3t);
    kvec_kernel<<<1, 256, 0, stream>>>(W_lin, b_lin, k_plus, k_minus, kp2, km2, bk);
    seg_kernel<<<1, 256, 0, stream>>>(num_atoms, Mm, Nn, offs, seg);
    num_kernel<<<(Nn + 7) / 8, 256, 0, stream>>>(e_z, seg, psi, kp2, km2, bk, numv, Nn);
    denom_kernel<<<(Mm + 7) / 8, 256, 0, stream>>>(numv, offs, num_atoms, denom, Mm);

    // dynamic LDS: X tile 64KB + W 128KB + v/a/pos staging (CDNA5: 320KB/WGP)
    size_t smemBytes = 65536 + 131072 + 2 * 256 * 4 + TM * 4 + TM * 4;
    mlp_kernel<<<(Nn + TM - 1) / TM, 512, smemBytes, stream>>>(
        numv, denom, seg, psi, v_plus, v_minus, act_alpha, act_beta,
        W1t, W2t, W3t, out, Nn);
}
